// TripletCorrelationalDetector_16518444220899
// MI455X (gfx1250) — matrix operand order repaired and checked by reading, not compiled, not run on hardware
//
#include <hip/hip_runtime.h>
#include <hip/hip_bf16.h>

typedef __attribute__((ext_vector_type(16))) _Float16 v16h;
typedef __attribute__((ext_vector_type(8)))  float    v8f;

// ---------------------------------------------------------------------------
// Implicit-GEMM convolution on WMMA (VALID, NCHW/OIHW), f16 mul / f32 acc.
//   M = 16 output channels, N = 4 x 16 flattened (b,oy,ox), K = CIN*KH*KW
// Each wave computes a 16x64 output tile (4 WMMAs per K-chunk, shared A).
// K-loop is unrolled by 2 with true double buffering: even chunks compute
// from buffer A while buffer B loads, odd chunks the reverse -- loads land
// directly in the registers the next compute reads (no cross-iteration moves),
// and load latency overlaps the WMMAs.
// Requires: P % 64 == 0, COUT % 16 == 0 (true for every conv in this net),
// and KH*KW == 16 (4x4) or KH*KW == 1 (pointwise).
// ---------------------------------------------------------------------------
template <int CIN, int KH, int KW, int S, bool RELU>
__global__ void __launch_bounds__(256, 1)
conv_fwd_wmma(const float* __restrict__ X, const float* __restrict__ W,
              const float* __restrict__ Bias, float* __restrict__ Y,
              int N, int H, int Wd, int COUT, int OH, int OW) {
  constexpr int KHW  = KH * KW;
  constexpr int KT   = CIN * KHW;
  constexpr bool FULL = (KT % 32) == 0;   // no K masking needed
  constexpr int NT   = 4;                 // 4 N-tiles per wave

  static_assert(KHW == 16 || KHW == 1, "kernel supports 4x4 or 1x1 only");

  const int lane = threadIdx.x & 31;
  const int wave = threadIdx.x >> 5;
  const int half = lane >> 4;
  const int l16  = lane & 15;

  const int P     = N * OH * OW;
  const int wtile = blockIdx.x * (blockDim.x >> 5) + wave;   // group of 4 ptiles
  const int cout0 = blockIdx.y * 16;
  if (wtile * (16 * NT) >= P) return;     // wave-uniform: EXEC stays all-ones

  const size_t chanHW = (size_t)H * Wd;

  // Per-lane base pointers for the 4 N columns (cin added per K-chunk)
  const float* xb[NT];
#pragma unroll
  for (int j = 0; j < NT; ++j) {
    int p = wtile * (16 * NT) + j * 16 + l16;           // always < P
    int b = p / (OH * OW); int r = p % (OH * OW);
    int oy = r / OW, ox = r % OW;
    xb[j] = X + ((size_t)b * CIN * H + (size_t)(oy * S)) * Wd + (size_t)(ox * S);
  }
  const float* wrow = W + (size_t)(cout0 + l16) * KT;   // A row m = l16

  // ---- loaders (clamped safe addresses; masks applied at convert time) ----
  auto loadA = [&](int kk, float4 (&A0)[2], float4 (&A1)[2]) {
#pragma unroll
    for (int g = 0; g < 2; ++g) {
      int kbase = kk + half * 16 + g * 8;
      bool av = FULL || (kbase < KT);
      const float4* ap = reinterpret_cast<const float4*>(wrow + (av ? kbase : 0));
      A0[g] = ap[0];
      A1[g] = ap[1];
    }
  };
  auto loadB = [&](int kk, float (&T)[NT][16]) {
    if constexpr (KHW == 16) {
      // Half-chunk of 16 consecutive k == full 4x4 patch of one input channel
      int cin = (kk >> 4) + half;
      size_t coff = (size_t)((FULL || cin < CIN) ? cin : 0) * chanHW;
#pragma unroll
      for (int j = 0; j < NT; ++j) {
        const float* xp = xb[j] + coff;
#pragma unroll
        for (int kh = 0; kh < 4; ++kh)
#pragma unroll
          for (int kw = 0; kw < 4; ++kw)
            T[j][kh * 4 + kw] = xp[(size_t)kh * Wd + kw];
      }
    } else {
      // 1x1 conv: 16 consecutive k == 16 channel-strided scalars
      int cin0 = kk + half * 16;
#pragma unroll
      for (int j = 0; j < NT; ++j)
#pragma unroll
        for (int e = 0; e < 16; ++e)
          T[j][e] = xb[j][(size_t)(cin0 + e) * chanHW];
    }
  };

  v8f acc[NT];
#pragma unroll
  for (int j = 0; j < NT; ++j) acc[j] = (v8f){};

  // ---- convert + WMMA for one chunk (masks recomputed from chunk base) ----
  auto compute = [&](int kc, float4 (&A0)[2], float4 (&A1)[2], float (&T)[NT][16]) {
    v16h va;
#pragma unroll
    for (int g = 0; g < 2; ++g) {
      float t[8] = {A0[g].x, A0[g].y, A0[g].z, A0[g].w,
                    A1[g].x, A1[g].y, A1[g].z, A1[g].w};
      if (!FULL) {
        float mA = ((kc + half * 16 + g * 8) < KT) ? 1.0f : 0.0f;
#pragma unroll
        for (int e = 0; e < 8; ++e) t[e] *= mA;
      }
#pragma unroll
      for (int e = 0; e < 8; ++e) va[g * 8 + e] = (_Float16)t[e];
    }
    float mB = 1.0f;
    if (!FULL && KHW == 16) mB = (((kc >> 4) + half) < CIN) ? 1.0f : 0.0f;
    v16h vb[NT];
#pragma unroll
    for (int j = 0; j < NT; ++j) {
#pragma unroll
      for (int e = 0; e < 16; ++e) {
        float x = T[j][e];
        if (!FULL) x *= mB;
        vb[j][e] = (_Float16)x;
      }
    }
#pragma unroll
    for (int j = 0; j < NT; ++j)
      acc[j] = __builtin_amdgcn_wmma_f32_16x16x32_f16(false, va, false, vb[j],
                                                      (short)0, acc[j], false, false);
  };

  // ---- double-buffered main loop, unrolled by 2 chunks ----
  float4 A0a[2], A1a[2], A0b[2], A1b[2];
  float tba[NT][16], tbb[NT][16];
  loadA(0, A0a, A1a);
  loadB(0, tba);

  for (int k0 = 0; k0 < KT; k0 += 64) {
    if (k0 + 32 < KT) {                    // prefetch odd chunk into buffer B
      loadA(k0 + 32, A0b, A1b);
      loadB(k0 + 32, tbb);
    }
    compute(k0, A0a, A1a, tba);            // compute even chunk from buffer A
    if (k0 + 64 < KT) {                    // prefetch next even chunk into A
      loadA(k0 + 64, A0a, A1a);
      loadB(k0 + 64, tba);
    }
    if (k0 + 32 < KT)
      compute(k0 + 32, A0b, A1b, tbb);     // compute odd chunk from buffer B
  }

  // ---- store: C/D layout VGPR r -> M = r + half*8, column n = l16 ----
#pragma unroll
  for (int j = 0; j < NT; ++j) {
    int p = wtile * (16 * NT) + j * 16 + l16;
    int b = p / (OH * OW); int r2 = p % (OH * OW);
    int oy = r2 / OW, ox = r2 % OW;
#pragma unroll
    for (int r = 0; r < 8; ++r) {
      int co = cout0 + r + half * 8;
      float v = acc[j][r] + Bias[co];
      if (RELU) v = fmaxf(v, 0.0f);
      Y[((size_t)b * COUT + co) * OH * OW + (size_t)oy * OW + ox] = v;
    }
  }
}

// ---------------------------------------------------------------------------
// Transposed conv (lhs_dilation = S, pad = K-1, kernel NOT flipped; jax
// conv_transpose default). W is OIHW: [COUT][CIN][KH][KW]. act: 0 none/1 relu/2 sigmoid
// ---------------------------------------------------------------------------
__global__ void deconv_kernel(const float* __restrict__ X, const float* __restrict__ W,
                              const float* __restrict__ Bias, float* __restrict__ Y,
                              int N, int CIN, int IH, int IW, int COUT,
                              int KH, int KW, int S, int OH, int OW, int act) {
  int idx = blockIdx.x * blockDim.x + threadIdx.x;
  int total = N * COUT * OH * OW;
  if (idx >= total) return;
  int ox = idx % OW; int t = idx / OW;
  int oy = t % OH;   t /= OH;
  int co = t % COUT; int b = t / COUT;
  float acc = Bias[co];
  for (int kh = 0; kh < KH; ++kh) {
    int ty = oy + kh - (KH - 1);
    if (ty < 0 || (ty % S)) continue;
    int iy = ty / S; if (iy >= IH) continue;
    for (int kw = 0; kw < KW; ++kw) {
      int tx = ox + kw - (KW - 1);
      if (tx < 0 || (tx % S)) continue;
      int ix = tx / S; if (ix >= IW) continue;
      const float* wp = W + (((size_t)co * CIN) * KH + kh) * KW + kw;
      const float* xp = X + (((size_t)b * CIN) * IH + iy) * IW + ix;
      for (int ci = 0; ci < CIN; ++ci)
        acc += wp[(size_t)ci * KH * KW] * xp[(size_t)ci * IH * IW];
    }
  }
  if (act == 1)      acc = fmaxf(acc, 0.0f);
  else if (act == 2) acc = 1.0f / (1.0f + __expf(-acc));
  Y[idx] = acc;
}

// ---------------------------------------------------------------------------
// Per-sample cross-correlation: out[b,oy,ox] = sum_{c,h,w} F[b,c,oy+h,ox+w]*A[b,c,h,w]
// One block per output element, LDS tree reduction.
// ---------------------------------------------------------------------------
__global__ void xcorr_kernel(const float* __restrict__ F, const float* __restrict__ A,
                             float* __restrict__ Y,
                             int C, int FH, int FW, int AH, int AW, int OH, int OW) {
  __shared__ float red[256];
  int o = blockIdx.x;
  int ox = o % OW; int t = o / OW; int oy = t % OH; int b = t / OH;
  const int K = C * AH * AW;
  const float* Fb = F + (size_t)b * C * FH * FW;
  const float* Ab = A + (size_t)b * C * AH * AW;
  float acc = 0.0f;
  for (int k = threadIdx.x; k < K; k += blockDim.x) {
    int c = k / (AH * AW); int r = k % (AH * AW);
    int h = r / AW, w = r % AW;
    acc += Fb[((size_t)c * FH + oy + h) * FW + (ox + w)] * Ab[k];
  }
  red[threadIdx.x] = acc;
  __syncthreads();
  for (int s = 128; s > 0; s >>= 1) {
    if ((int)threadIdx.x < s) red[threadIdx.x] += red[threadIdx.x + s];
    __syncthreads();
  }
  if (threadIdx.x == 0) Y[o] = red[0];
}

__global__ void bcast_anchor(const float* __restrict__ a, float* __restrict__ out,
                             int per, int total) {
  int i = blockIdx.x * blockDim.x + threadIdx.x;
  if (i < total) out[i] = a[i % per];
}

__global__ void reparam_kernel(const float* __restrict__ mu, const float* __restrict__ lv,
                               const float* __restrict__ eps, float* __restrict__ z, int n) {
  int i = blockIdx.x * blockDim.x + threadIdx.x;
  if (i < n) z[i] = mu[i] + eps[i] * __expf(0.5f * lv[i]);
}

// ---------------------------------------------------------------------------
// Orchestration
// ---------------------------------------------------------------------------
template <int CIN, int KH, int KW, int S, bool RELU>
static void launch_conv(const float* X, const float* W, const float* B, float* Y,
                        int N, int H, int Wd, int COUT, int OH, int OW, hipStream_t stream) {
  int P = N * OH * OW;                 // always a multiple of 64 here
  int wtiles = P / 64;
  dim3 grid((wtiles + 7) / 8, COUT / 16);
  conv_fwd_wmma<CIN, KH, KW, S, RELU><<<grid, 256, 0, stream>>>(X, W, B, Y, N, H, Wd, COUT, OH, OW);
}

extern "C" void kernel_launch(void* const* d_in, const int* in_sizes, int n_in,
                              void* d_out, int out_size, void* d_ws, size_t ws_size,
                              hipStream_t stream) {
  const float* pos      = (const float*)d_in[0];
  const float* neg      = (const float*)d_in[1];
  const float* pos_crop = (const float*)d_in[2];
  const float* eps      = (const float*)d_in[3];
  const float* anc_crop = (const float*)d_in[4];
  const float* ew1 = (const float*)d_in[5];  const float* eb1 = (const float*)d_in[6];
  const float* ew2 = (const float*)d_in[7];  const float* eb2 = (const float*)d_in[8];
  const float* ew3 = (const float*)d_in[9];  const float* eb3 = (const float*)d_in[10];
  const float* mw  = (const float*)d_in[11]; const float* mb  = (const float*)d_in[12];
  const float* lw  = (const float*)d_in[13]; const float* lb  = (const float*)d_in[14];
  const float* dw0 = (const float*)d_in[15]; const float* db0 = (const float*)d_in[16];
  const float* dw1 = (const float*)d_in[17]; const float* db1 = (const float*)d_in[18];
  const float* dw2 = (const float*)d_in[19]; const float* db2 = (const float*)d_in[20];
  const float* dw3 = (const float*)d_in[21]; const float* db3 = (const float*)d_in[22];
  float* out = (float*)d_out;
  float* ws  = (float*)d_ws;

  constexpr int B = 64;
  // ---- output layout (floats, concat in return order) ----
  constexpr size_t SZ_IMG = (size_t)B * 3 * 127 * 127;   // anchor / recon
  constexpr size_t SZ_CX  = (size_t)B * 17 * 17;
  constexpr size_t SZ_MU  = (size_t)B * 32 * 14 * 14;
  float* o_anchor = out;
  float* o_cxp    = out + SZ_IMG;
  float* o_cxn    = o_cxp + SZ_CX;
  float* o_recon  = o_cxn + SZ_CX;
  float* o_mu     = o_recon + SZ_IMG;
  float* o_lv     = o_mu + SZ_MU;

  // ---- workspace layout (floats) ----
  constexpr size_t CAP_A = (size_t)B * 32 * 126 * 126;   // big conv1 out
  constexpr size_t CAP_B = (size_t)B * 64 * 62 * 62;     // conv2 out
  constexpr size_t SZ_F30 = (size_t)B * 64 * 30 * 30;
  constexpr size_t SZ_F14 = (size_t)B * 64 * 14 * 14;
  float* bufA  = ws;
  float* bufB  = ws + CAP_A;
  float* posf  = bufB + CAP_B;
  float* negf  = posf + SZ_F30;
  float* ancf  = negf + SZ_F30;
  // decoder small buffers reuse bufB after it is dead
  float* zbuf  = bufB;
  float* d0buf = bufB + SZ_MU;
  float* d1buf = d0buf + SZ_F14;

  // 1) broadcast anchor -> output (also serves as encoder input)
  {
    int per = 3 * 127 * 127, total = (int)SZ_IMG;
    bcast_anchor<<<(total + 255) / 256, 256, 0, stream>>>(anc_crop, o_anchor, per, total);
  }

  // 2) enc(pos) -> posf
  launch_conv<3, 4, 4, 2, true >(pos,  ew1, eb1, bufA, B, 255, 255, 32, 126, 126, stream);
  launch_conv<32, 4, 4, 2, true>(bufA, ew2, eb2, bufB, B, 126, 126, 64, 62, 62, stream);
  launch_conv<64, 4, 4, 2, true>(bufB, ew3, eb3, posf, B, 62, 62, 64, 30, 30, stream);
  // 3) enc(neg) -> negf
  launch_conv<3, 4, 4, 2, true >(neg,  ew1, eb1, bufA, B, 255, 255, 32, 126, 126, stream);
  launch_conv<32, 4, 4, 2, true>(bufA, ew2, eb2, bufB, B, 126, 126, 64, 62, 62, stream);
  launch_conv<64, 4, 4, 2, true>(bufB, ew3, eb3, negf, B, 62, 62, 64, 30, 30, stream);
  // 4) enc(anchor) -> ancf   (reads broadcast anchor from d_out)
  launch_conv<3, 4, 4, 2, true >(o_anchor, ew1, eb1, bufA, B, 127, 127, 32, 62, 62, stream);
  launch_conv<32, 4, 4, 2, true>(bufA, ew2, eb2, bufB, B, 62, 62, 64, 30, 30, stream);
  launch_conv<64, 4, 4, 2, true>(bufB, ew3, eb3, ancf, B, 30, 30, 64, 14, 14, stream);
  // 5) enc(pos_crop) -> hc3 (bufA), then VAE heads
  launch_conv<3, 4, 4, 2, true >(pos_crop, ew1, eb1, bufA, B, 127, 127, 32, 62, 62, stream);
  launch_conv<32, 4, 4, 2, true>(bufA, ew2, eb2, bufB, B, 62, 62, 64, 30, 30, stream);
  launch_conv<64, 4, 4, 2, true>(bufB, ew3, eb3, bufA, B, 30, 30, 64, 14, 14, stream); // hc3
  launch_conv<64, 1, 1, 1, false>(bufA, mw, mb, o_mu, B, 14, 14, 32, 14, 14, stream);
  launch_conv<64, 1, 1, 1, false>(bufA, lw, lb, o_lv, B, 14, 14, 32, 14, 14, stream);

  // 6) reparameterize: z = mu + eps*exp(0.5*logvar)   (bufB dead -> reuse)
  {
    int n = (int)SZ_MU;
    reparam_kernel<<<(n + 255) / 256, 256, 0, stream>>>(o_mu, o_lv, eps, zbuf, n);
  }
  // 7) decoder head: 1x1 conv 32->64 + relu
  launch_conv<32, 1, 1, 1, true>(zbuf, dw0, db0, d0buf, B, 14, 14, 64, 14, 14, stream);
  // 8) deconv1 64->32, 4x4 s2, 14->30, relu
  {
    int total = B * 32 * 30 * 30;
    deconv_kernel<<<(total + 255) / 256, 256, 0, stream>>>(d0buf, dw1, db1, d1buf,
                                                           B, 64, 14, 14, 32, 4, 4, 2, 30, 30, 1);
  }
  // 9) deconv2 32->16, 4x4 s2, 30->62, relu   (bufA dead -> reuse)
  {
    int total = B * 16 * 62 * 62;
    deconv_kernel<<<(total + 255) / 256, 256, 0, stream>>>(d1buf, dw2, db2, bufA,
                                                           B, 32, 30, 30, 16, 4, 4, 2, 62, 62, 1);
  }
  // 10) deconv3 16->3, 5x5 s2, 62->127, sigmoid -> recon
  {
    int total = (int)SZ_IMG;
    deconv_kernel<<<(total + 255) / 256, 256, 0, stream>>>(bufA, dw3, db3, o_recon,
                                                           B, 16, 62, 62, 3, 5, 5, 2, 127, 127, 2);
  }
  // 11) cross-correlations
  {
    int blocks = B * 17 * 17;
    xcorr_kernel<<<blocks, 256, 0, stream>>>(posf, ancf, o_cxp, 64, 30, 30, 14, 14, 17, 17);
    xcorr_kernel<<<blocks, 256, 0, stream>>>(negf, ancf, o_cxn, 64, 30, 30, 14, 14, 17, 17);
  }
}
